// MultiHeadSelfAttention_18511309046034
// MI455X (gfx1250) — matrix operand order repaired and checked by reading, not compiled, not run on hardware
//
#include <hip/hip_runtime.h>
#include <hip/hip_bf16.h>
#include <math.h>

// Problem constants (fixed by the reference)
#define B_DIM    4
#define L_DIM    2048
#define D_MODEL  1024
#define N_HEADS  16
#define HEAD_DIM 64
#define M_ROWS   (B_DIM * L_DIM)   // 8192

typedef __bf16 bf16;
typedef __attribute__((ext_vector_type(16))) __bf16 v16bf;
typedef __attribute__((ext_vector_type(8)))  float  v8f;

// 32-byte fragment viewed as two 16-byte quads -> ds_load_b128/global_load_b128
union FragB16 { v16bf v; uint4 q[2]; };

// CDNA5 async global->LDS copy path (ASYNCcnt) if the toolchain exposes it.
#if __has_builtin(__builtin_amdgcn_global_load_async_to_lds_b128) && \
    __has_builtin(__builtin_amdgcn_s_wait_asynccnt)
#define USE_ASYNC_LDS 1
// Builtin signature (from hipcc diagnostic): 16-byte int vector pointers,
// global (addrspace 1) source first, LDS (addrspace 3) destination second.
typedef int async_v4i __attribute__((vector_size(16)));
#define GLOBAL_AS(p) ((__attribute__((address_space(1))) async_v4i*)(p))
#define LDS_AS(p)    ((__attribute__((address_space(3))) async_v4i*)(p))
#endif

__device__ __forceinline__ void stage16(const void* g, void* l) {
#ifdef USE_ASYNC_LDS
  __builtin_amdgcn_global_load_async_to_lds_b128(GLOBAL_AS(g), LDS_AS(l), 0, 0);
#else
  *(uint4*)l = *(const uint4*)g;
#endif
}

__device__ __forceinline__ void stage_wait() {
#ifdef USE_ASYNC_LDS
  __builtin_amdgcn_s_wait_asynccnt(0);
#endif
}

__device__ __forceinline__ bf16 f2bf(float f) {
  union { float f; unsigned u; } in; in.f = f;
  unsigned r = (in.u + 0x7FFFu + ((in.u >> 16) & 1u)) >> 16;  // RNE
  union { unsigned short s; bf16 b; } out; out.s = (unsigned short)r;
  return out.b;
}

__device__ __forceinline__ v8f wmma_bf16(v16bf a, v16bf b, v8f c) {
  return __builtin_amdgcn_wmma_f32_16x16x32_bf16(false, a, false, b, (short)0, c,
                                                 false, false);
}

// ---------------------------------------------------------------------------
// Kernel 1: fp32 -> bf16 TRANSPOSE convert for weights: Wt[n][k] = W[k][n].
// Tiled 32x32 through LDS so both global read and write stay coalesced.
// One-time cost; makes all GEMM B-staging a pure b128 memcpy.
// ---------------------------------------------------------------------------
__global__ __launch_bounds__(256)
void transpose_bf16_kernel(const float* __restrict__ src,  // K x N (= D x D)
                           bf16* __restrict__ dst) {       // N x K
  __shared__ float tile[32][33];
  const int n0 = blockIdx.x * 32;
  const int k0 = blockIdx.y * 32;
  const int tx = threadIdx.x & 31;
  const int ty = threadIdx.x >> 5;      // 0..7
  #pragma unroll
  for (int i = 0; i < 32; i += 8)
    tile[ty + i][tx] = src[(size_t)(k0 + ty + i) * D_MODEL + n0 + tx];
  __syncthreads();
  #pragma unroll
  for (int i = 0; i < 32; i += 8)
    dst[(size_t)(n0 + ty + i) * D_MODEL + k0 + tx] = f2bf(tile[tx][ty + i]);
}

// ---------------------------------------------------------------------------
// Kernel 2: fused LayerNorm -> bf16 h (one block per row of 1024)
// ---------------------------------------------------------------------------
__global__ __launch_bounds__(256)
void layernorm_bf16_kernel(const float* __restrict__ x,
                           const float* __restrict__ gamma,
                           const float* __restrict__ beta,
                           bf16* __restrict__ h) {
  const int row = blockIdx.x;
  const float* xr = x + (size_t)row * D_MODEL;
  bf16* hr = h + (size_t)row * D_MODEL;

  float s = 0.f, s2 = 0.f;
  for (int i = threadIdx.x; i < D_MODEL; i += 256) {
    float v = xr[i]; s += v; s2 += v * v;
  }
  #pragma unroll
  for (int off = 16; off > 0; off >>= 1) {
    s  += __shfl_xor(s,  off, 32);
    s2 += __shfl_xor(s2, off, 32);
  }
  __shared__ float red0[8], red1[8];
  const int wid = threadIdx.x >> 5;
  if ((threadIdx.x & 31) == 0) { red0[wid] = s; red1[wid] = s2; }
  __syncthreads();
  float ts = 0.f, ts2 = 0.f;
  #pragma unroll
  for (int w = 0; w < 8; ++w) { ts += red0[w]; ts2 += red1[w]; }
  const float mu  = ts  * (1.0f / D_MODEL);
  const float var = ts2 * (1.0f / D_MODEL) - mu * mu;
  const float rs  = rsqrtf(var + 1e-5f);
  for (int i = threadIdx.x; i < D_MODEL; i += 256)
    hr[i] = f2bf((xr[i] - mu) * rs * gamma[i] + beta[i]);
}

// ---------------------------------------------------------------------------
// Kernel 3: tiled bf16 WMMA GEMM.  out = (A @ W + bias) * scale [+ residual]
//   A:  M x K bf16 row-major
//   Bt: N x K bf16 row-major (TRANSPOSED weights -> contiguous-K fragments)
//   mode 0: bf16 scatter -> (B, H, L, Hd)      (Q, K projections)
//   mode 2: bf16 scatter -> (B, H, Hd, L)      (V projection, transposed)
//   mode 1: fp32 row-major + residual          (output projection)
// Block: 256 threads (8 waves), 64x128 tile, K-step 32, 4 WMMA / wave / step.
// All staging is b128 global->LDS (async path when available).
// ---------------------------------------------------------------------------
#define TM 64
#define TN 128
#define TK 32
#define LSTR 40   // LDS row stride in bf16 (80 B: 16-B aligned, bank-spread)

__global__ __launch_bounds__(256)
void gemm_bf16_kernel(const bf16* __restrict__ A,
                      const bf16* __restrict__ Bt,
                      const float* __restrict__ bias,
                      const float* __restrict__ residual,
                      void* __restrict__ out,
                      int M, int N, int K,
                      float scale, int mode) {
  __shared__ __align__(16) bf16 As[TM][LSTR];   // As[row][k]
  __shared__ __align__(16) bf16 Bs[TN][LSTR];   // Bs[col][k] (K contiguous)

  const int tid  = threadIdx.x;
  const int lane = tid & 31;
  const int wave = tid >> 5;     // 0..7
  const int wr   = wave >> 1;    // 0..3  (16-row group)
  const int wc   = wave & 1;     // 0..1  (64-col group)
  const int m0   = blockIdx.x * TM;
  const int n0   = blockIdx.y * TN;

  // staging assignments (all 16-B vector copies)
  const int ar = tid >> 2,  ac = (tid & 3) * 8;    // A: 16 B / thread
  const int bcol = tid >> 1, bk = (tid & 1) * 16;  // B: 32 B / thread

  const int cl  = lane & 15;
  const int kbA = (lane >> 4) << 3;   // 0 or 8
  const int kbB = (lane >> 4) << 4;   // 0 or 16

  v8f acc[4];
  #pragma unroll
  for (int t = 0; t < 4; ++t) acc[t] = (v8f){};

  for (int k0 = 0; k0 < K; k0 += TK) {
    stage16(&A[(size_t)(m0 + ar) * K + k0 + ac],        &As[ar][ac]);
    stage16(&Bt[(size_t)(n0 + bcol) * K + k0 + bk],     &Bs[bcol][bk]);
    stage16(&Bt[(size_t)(n0 + bcol) * K + k0 + bk + 8], &Bs[bcol][bk + 8]);
    stage_wait();
    __syncthreads();

    // A fragment: two contiguous 16-B runs per lane (ds_load_b128 x2)
    FragB16 a;
    {
      const int row = (wr << 4) + cl;
      a.q[0] = *(const uint4*)&As[row][kbA];
      a.q[1] = *(const uint4*)&As[row][16 + kbA];
    }
    // 4 B fragments: same K runs as A (ds_load_b128 x2 each)
    #pragma unroll
    for (int t = 0; t < 4; ++t) {
      FragB16 b;
      const bf16* p = &Bs[(wc << 6) + (t << 4) + cl][kbA];
      b.q[0] = *(const uint4*)p;
      b.q[1] = *(const uint4*)(p + 16);
      acc[t] = wmma_bf16(a.v, b.v, acc[t]);
    }
    __syncthreads();
  }

  // store: C layout -> lane<16 rows 0..7, lane>=16 rows 8..15; col = lane&15
  const int rbase = (lane >> 4) << 3;
  #pragma unroll
  for (int t = 0; t < 4; ++t) {
    #pragma unroll
    for (int r = 0; r < 8; ++r) {
      const int gr = m0 + (wr << 4) + rbase + r;
      const int gc = n0 + (wc << 6) + (t << 4) + cl;
      const float v = (acc[t][r] + bias[gc]) * scale;
      if (mode == 1) {
        const size_t o = (size_t)gr * N + gc;
        ((float*)out)[o] = v + residual[o];
      } else {
        const int bb = gr >> 11;            // / L_DIM
        const int l  = gr & (L_DIM - 1);
        const int hh = gc >> 6;             // / HEAD_DIM
        const int hd = gc & (HEAD_DIM - 1);
        const size_t head = (size_t)(bb * N_HEADS + hh);
        const size_t off = (mode == 0)
            ? (head * L_DIM + l) * HEAD_DIM + hd      // (B,H,L,Hd)
            : (head * HEAD_DIM + hd) * L_DIM + l;     // (B,H,Hd,L)
        ((bf16*)out)[off] = f2bf(v);
      }
    }
  }
}

// ---------------------------------------------------------------------------
// Kernel 4: flash attention. One wave per (b, h, 16-query tile).
//   Q pre-scaled by 1/sqrt(Hd). 32 keys / iteration:
//   S(16x32) = 4 WMMAs, online softmax, P -> bf16 via LDS relayout,
//   O += P @ V = 4 WMMAs (V stored (B,H,Hd,L) so fragments are b128 loads).
// ---------------------------------------------------------------------------
__global__ __launch_bounds__(32)
void attention_kernel(const bf16* __restrict__ Q,    // (B,H,L,Hd)
                      const bf16* __restrict__ Km,   // (B,H,L,Hd)
                      const bf16* __restrict__ Vt,   // (B,H,Hd,L)
                      const unsigned char* __restrict__ mask,  // (B,L) bool
                      bf16* __restrict__ attn_out) {           // (B,L,D)
  const int lane = threadIdx.x & 31;
  const int q0   = blockIdx.x << 4;
  const int h    = blockIdx.y;
  const int b    = blockIdx.z;

  const bf16* Qh  = Q  + ((size_t)(b * N_HEADS + h)) * L_DIM * HEAD_DIM;
  const bf16* Kh  = Km + ((size_t)(b * N_HEADS + h)) * L_DIM * HEAD_DIM;
  const bf16* Vth = Vt + ((size_t)(b * N_HEADS + h)) * HEAD_DIM * L_DIM;

  const int cl  = lane & 15;
  const int kbA = (lane >> 4) << 3;   // 0 or 8
  const int kbB = (lane >> 4) << 4;   // 0 or 16

  // Q tile: two A fragments (K windows [0,32), [32,64)), b128 loads
  FragB16 aq0, aq1;
  {
    const bf16* qr = Qh + (size_t)(q0 + cl) * HEAD_DIM;
    aq0.q[0] = *(const uint4*)(qr + kbA);
    aq0.q[1] = *(const uint4*)(qr + 16 + kbA);
    aq1.q[0] = *(const uint4*)(qr + 32 + kbA);
    aq1.q[1] = *(const uint4*)(qr + 48 + kbA);
  }

  v8f o[4];
  #pragma unroll
  for (int d = 0; d < 4; ++d) o[d] = (v8f){};
  float mrow[8], lsum[8];
  #pragma unroll
  for (int r = 0; r < 8; ++r) { mrow[r] = -INFINITY; lsum[r] = 0.f; }

  __shared__ __align__(16) bf16 Plds[16][LSTR];

  for (int j0 = 0; j0 < L_DIM; j0 += 32) {
    // prefetch next key/value tiles while computing this one
    if (j0 + 32 < L_DIM) {
      __builtin_prefetch(Kh + (size_t)(j0 + 32 + cl) * HEAD_DIM, 0, 1);
      __builtin_prefetch(Vth + (size_t)((lane & 3) << 4) * L_DIM + j0 + 32, 0, 1);
    }

    // ---- S = Q @ K^T over 32 keys (two 16x16 C fragments) ----
    v8f s[2]; s[0] = (v8f){}; s[1] = (v8f){};
    #pragma unroll
    for (int t = 0; t < 2; ++t) {
      const bf16* krow = Kh + (size_t)(j0 + (t << 4) + cl) * HEAD_DIM;
      FragB16 bk0, bk1;                 // B[kk][key] = K[key][kk], contiguous
      bk0.q[0] = *(const uint4*)(krow + kbA);
      bk0.q[1] = *(const uint4*)(krow + 16 + kbA);
      bk1.q[0] = *(const uint4*)(krow + 32 + kbA);
      bk1.q[1] = *(const uint4*)(krow + 48 + kbA);
      s[t] = wmma_bf16(aq0.v, bk0.v, s[t]);
      s[t] = wmma_bf16(aq1.v, bk1.v, s[t]);
    }

    const float mk0 = mask[b * L_DIM + j0 + cl]      ? 0.f : -3.0e38f;
    const float mk1 = mask[b * L_DIM + j0 + 16 + cl] ? 0.f : -3.0e38f;

    // ---- online softmax ----
    float p0[8], p1[8];
    #pragma unroll
    for (int r = 0; r < 8; ++r) {
      float s0 = s[0][r] + mk0;
      float s1 = s[1][r] + mk1;
      float tmax = fmaxf(s0, s1);
      #pragma unroll
      for (int off = 1; off < 16; off <<= 1)
        tmax = fmaxf(tmax, __shfl_xor(tmax, off, 32));
      const float nm   = fmaxf(mrow[r], tmax);
      const float corr = __expf(mrow[r] - nm);
      mrow[r] = nm;
      p0[r] = __expf(s0 - nm);
      p1[r] = __expf(s1 - nm);
      float rs = p0[r] + p1[r];
      #pragma unroll
      for (int off = 1; off < 16; off <<= 1)
        rs += __shfl_xor(rs, off, 32);
      lsum[r] = lsum[r] * corr + rs;
      o[0][r] *= corr; o[1][r] *= corr; o[2][r] *= corr; o[3][r] *= corr;
    }

    // ---- P: C-layout f32 -> LDS -> A-layout bf16 ----
    {
      const int rb = (lane >> 4) << 3;
      #pragma unroll
      for (int r = 0; r < 8; ++r) {
        Plds[rb + r][cl]      = f2bf(p0[r]);
        Plds[rb + r][16 + cl] = f2bf(p1[r]);
      }
    }
    __syncthreads();
    FragB16 ap;
    ap.q[0] = *(const uint4*)&Plds[cl][kbA];
    ap.q[1] = *(const uint4*)&Plds[cl][16 + kbA];

    // ---- O += P @ V (V transposed: contiguous b128 fragments) ----
    #pragma unroll
    for (int d = 0; d < 4; ++d) {
      const bf16* vcol = Vth + (size_t)((d << 4) + cl) * L_DIM + j0 + kbA;
      FragB16 bv;
      bv.q[0] = *(const uint4*)vcol;
      bv.q[1] = *(const uint4*)(vcol + 16);
      o[d] = wmma_bf16(ap.v, bv.v, o[d]);
    }
    __syncthreads();
  }

  // ---- normalize and store to (B, L, D) bf16 ----
  const int rb = (lane >> 4) << 3;
  #pragma unroll
  for (int r = 0; r < 8; ++r) {
    const float inv = 1.0f / lsum[r];
    const int l = q0 + rb + r;
    bf16* dst = attn_out + ((size_t)b * L_DIM + l) * D_MODEL + h * HEAD_DIM;
    #pragma unroll
    for (int d = 0; d < 4; ++d)
      dst[(d << 4) + cl] = f2bf(o[d][r] * inv);
  }
}

// ---------------------------------------------------------------------------
// Host-side orchestration
// ---------------------------------------------------------------------------
extern "C" void kernel_launch(void* const* d_in, const int* in_sizes, int n_in,
                              void* d_out, int out_size, void* d_ws, size_t ws_size,
                              hipStream_t stream) {
  (void)in_sizes; (void)n_in; (void)out_size; (void)ws_size;
  const float* x     = (const float*)d_in[0];
  const unsigned char* mask = (const unsigned char*)d_in[1];
  const float* gamma = (const float*)d_in[2];
  const float* beta  = (const float*)d_in[3];
  const float* Wq    = (const float*)d_in[4];
  const float* bq    = (const float*)d_in[5];
  const float* Wk    = (const float*)d_in[6];
  const float* bk    = (const float*)d_in[7];
  const float* Wv    = (const float*)d_in[8];
  const float* bv    = (const float*)d_in[9];
  const float* Wo    = (const float*)d_in[10];
  const float* bo    = (const float*)d_in[11];
  float* out = (float*)d_out;

  // workspace layout (bf16 buffers)
  char* ws = (char*)d_ws;
  bf16* hbf   = (bf16*)(ws);                                  // 8192x1024
  bf16* wqb   = (bf16*)(ws + (size_t)16 * 1024 * 1024);       // W^T, 1024x1024
  bf16* wkb   = wqb + (size_t)D_MODEL * D_MODEL;
  bf16* wvb   = wkb + (size_t)D_MODEL * D_MODEL;
  bf16* wob   = wvb + (size_t)D_MODEL * D_MODEL;
  bf16* Qb    = wob + (size_t)D_MODEL * D_MODEL;              // (B,H,L,Hd)
  bf16* Kb    = Qb  + (size_t)M_ROWS * D_MODEL;               // (B,H,L,Hd)
  bf16* Vb    = Kb  + (size_t)M_ROWS * D_MODEL;               // (B,H,Hd,L)
  bf16* attnb = Vb  + (size_t)M_ROWS * D_MODEL;               // (B,L,D)

  dim3 tgrid(D_MODEL / 32, D_MODEL / 32);
  transpose_bf16_kernel<<<tgrid, 256, 0, stream>>>(Wq, wqb);
  transpose_bf16_kernel<<<tgrid, 256, 0, stream>>>(Wk, wkb);
  transpose_bf16_kernel<<<tgrid, 256, 0, stream>>>(Wv, wvb);
  transpose_bf16_kernel<<<tgrid, 256, 0, stream>>>(Wo, wob);

  layernorm_bf16_kernel<<<M_ROWS, 256, 0, stream>>>(x, gamma, beta, hbf);

  dim3 ggrid(M_ROWS / TM, D_MODEL / TN);
  const float qscale = 0.125f;   // 1/sqrt(64) folded into Q (incl. bias)
  gemm_bf16_kernel<<<ggrid, 256, 0, stream>>>(hbf, wqb, bq, nullptr, Qb,
                                              M_ROWS, D_MODEL, D_MODEL, qscale, 0);
  gemm_bf16_kernel<<<ggrid, 256, 0, stream>>>(hbf, wkb, bk, nullptr, Kb,
                                              M_ROWS, D_MODEL, D_MODEL, 1.0f, 0);
  gemm_bf16_kernel<<<ggrid, 256, 0, stream>>>(hbf, wvb, bv, nullptr, Vb,
                                              M_ROWS, D_MODEL, D_MODEL, 1.0f, 2);

  dim3 agrid(L_DIM / 16, N_HEADS, B_DIM);
  attention_kernel<<<agrid, 32, 0, stream>>>(Qb, Kb, Vb, mask, attnb);

  gemm_bf16_kernel<<<ggrid, 256, 0, stream>>>(attnb, wob, bo, x, out,
                                              M_ROWS, D_MODEL, D_MODEL, 1.0f, 1);
}
